// NeuromorphicMemory_50964081934587
// MI455X (gfx1250) — matrix operand order repaired and checked by reading, not compiled, not run on hardware
//
#include <hip/hip_runtime.h>
#include <hip/hip_bf16.h>
#include <math.h>

#define HSZ 2048
#define H2  1024
#define NB  16384
#define THRV 0.7f
#define NEGF -1e30f

typedef __attribute__((ext_vector_type(16))) __bf16 v16bf;
typedef __attribute__((ext_vector_type(8)))  __bf16 v8bf;
typedef __attribute__((ext_vector_type(4)))  __bf16 v4bf;
typedef __attribute__((ext_vector_type(8)))  float  v8f;

// ---------------------------------------------------------------------------
// k0: convert+transpose W_e1 [K=2048][N=1024] f32 -> B1 bf16 [N=1024][K=2048]
//     (B-fragment friendly: a lane's 16 consecutive K values are one 32B load)
//     Also clears the hasEvents flag for this launch.
// ---------------------------------------------------------------------------
__global__ void __launch_bounds__(256)
k0_prep(const float* __restrict__ We1, __bf16* __restrict__ B1, int* hasEvents) {
  if (blockIdx.x == 0 && threadIdx.x == 0) *hasEvents = 0;
  int t = blockIdx.x * 256 + threadIdx.x;          // 2048*1024 elements
  int k = t & (HSZ - 1);
  int n = t >> 11;
  if (n < H2) B1[(size_t)n * HSZ + k] = (__bf16)We1[(size_t)k * H2 + n];
}

// ---------------------------------------------------------------------------
// k1: fused detector:  h = relu(x @ W_e1 + b_e1); s = sigmoid(h @ W_e2 + b_e2)
//     WMMA bf16 GEMM. 16 rows/block, 8 waves x (8 tiles of 16 cols).
//     A tile staged to LDS as bf16 in two K-phases (barrier-free inner loop).
// ---------------------------------------------------------------------------
__global__ void __launch_bounds__(256)
k1_detect(const float* __restrict__ x, const __bf16* __restrict__ B1,
          const float* __restrict__ b_e1, const float* __restrict__ We2,
          const float* __restrict__ b_e2, const float* __restrict__ importance,
          float* __restrict__ logits, int* __restrict__ evflags,
          int* __restrict__ hasEvents) {
  constexpr int KP   = 1024;       // K per staging phase
  constexpr int LSTR = KP + 8;     // padded LDS row stride (bank-conflict free)
  __shared__ __bf16 At[16 * LSTR]; // ~33 KB
  __shared__ float  rowacc[16];

  const int tid  = threadIdx.x;
  const int lane = tid & 31;
  const int wv   = tid >> 5;                 // wave 0..7
  const int rowbase = blockIdx.x * 16;

  v8f c[8];
#pragma unroll
  for (int t = 0; t < 8; ++t)
#pragma unroll
    for (int i = 0; i < 8; ++i) c[t][i] = 0.0f;

  // WMMA 16-bit A layout (contiguous, matches B): lane<16 -> K 0..15; lane>=16 -> K 16..31
  const int m    = lane & 15;
  const int khA  = (lane >> 4) * 16;
  const int khB  = (lane >> 4) * 16;         // B: lanes 0-15 K 0..15, 16-31 K 16..31
  const int ncol = lane & 15;
  const int cbase = wv * 128;

  for (int ph = 0; ph < 2; ++ph) {
    const int kb = ph * KP;
    __syncthreads();                         // previous phase's LDS reads done
    // stage 16 x 1024 floats -> bf16 LDS: 256 threads x 16 iters x float4
#pragma unroll 4
    for (int i = 0; i < 16; ++i) {
      int idx = (i * 256 + tid) * 4;
      int row = idx >> 10;
      int col = idx & (KP - 1);
      float4 xv = *(const float4*)(x + (size_t)(rowbase + row) * HSZ + kb + col);
      v4bf o;
      o[0] = (__bf16)xv.x; o[1] = (__bf16)xv.y;
      o[2] = (__bf16)xv.z; o[3] = (__bf16)xv.w;
      *(v4bf*)(&At[row * LSTR + col]) = o;   // ds_store_b64
    }
    __syncthreads();

    for (int kk = 0; kk < KP; kk += 32) {
      const int k0 = kb + kk;
      // A fragment from LDS (two contiguous 16B halves)
      v8bf alo = *(const v8bf*)(&At[m * LSTR + kk + khA]);
      v8bf ahi = *(const v8bf*)(&At[m * LSTR + kk + khA + 8]);
      v16bf a;
#pragma unroll
      for (int i = 0; i < 8; ++i) { a[i] = alo[i]; a[i + 8] = ahi[i]; }

      // issue ALL 8 B-fragment loads (each one contiguous 32B), then WMMA
      const __bf16* bp0 = B1 + (size_t)(cbase + ncol) * HSZ + k0 + khB;
      __builtin_prefetch(bp0 + 128, 0, 1);   // speculative, next k-chunks
      v16bf bf[8];
#pragma unroll
      for (int t = 0; t < 8; ++t)
        bf[t] = *(const v16bf*)(bp0 + (size_t)t * 16 * HSZ);
#pragma unroll
      for (int t = 0; t < 8; ++t)
        c[t] = __builtin_amdgcn_wmma_f32_16x16x32_bf16(
            false, a, false, bf[t], (short)0, c[t], false, false);
    }
  }

  // epilogue: +b_e1, relu, scale by W_e2 column weight, reduce over N
  // C layout: c[t][r] -> row (r + 8*(lane>>4)), col (cbase + t*16 + ncol)
  float part[8];
#pragma unroll
  for (int r = 0; r < 8; ++r) part[r] = 0.0f;
#pragma unroll
  for (int t = 0; t < 8; ++t) {
    int ng = cbase + t * 16 + ncol;
    float w2 = We2[ng];
    float bb = b_e1[ng];
#pragma unroll
    for (int r = 0; r < 8; ++r) {
      float h = fmaxf(c[t][r] + bb, 0.0f);
      part[r] += h * w2;
    }
  }

  __syncthreads();
  if (tid < 16) rowacc[tid] = 0.0f;
  __syncthreads();
  const int rb = (lane >> 4) * 8;
#pragma unroll
  for (int r = 0; r < 8; ++r)
    __hip_atomic_fetch_add(&rowacc[rb + r], part[r],
                           __ATOMIC_RELAXED, __HIP_MEMORY_SCOPE_WORKGROUP);
  __syncthreads();

  if (tid < 16) {
    int row = rowbase + tid;
    float s = 1.0f / (1.0f + __expf(-(rowacc[tid] + b_e2[0])));
    int ev = (s > THRV) ? 1 : 0;
    logits[row]  = ev ? importance[row] : NEGF;
    evflags[row] = ev;
    if (ev) *hasEvents = 1;                    // monotonic flag, race-safe
  }
}

// ---------------------------------------------------------------------------
// k2: masked-softmax stats (max, 1/sum) over 16384 logits — single block
// ---------------------------------------------------------------------------
__global__ void __launch_bounds__(1024)
k2_stats(const float* __restrict__ logits, float* __restrict__ stats) {
  __shared__ float red[1024];
  int tid = threadIdx.x;
  float m = NEGF;
  for (int i = tid; i < NB; i += 1024) m = fmaxf(m, logits[i]);
  red[tid] = m; __syncthreads();
  for (int s = 512; s > 0; s >>= 1) {
    if (tid < s) red[tid] = fmaxf(red[tid], red[tid + s]);
    __syncthreads();
  }
  float M = red[0];
  __syncthreads();
  float sum = 0.0f;
  for (int i = tid; i < NB; i += 1024) sum += __expf(logits[i] - M);
  red[tid] = sum; __syncthreads();
  for (int s = 512; s > 0; s >>= 1) {
    if (tid < s) red[tid] += red[tid + s];
    __syncthreads();
  }
  if (tid == 0) { stats[0] = M; stats[1] = 1.0f / red[0]; }
}

// ---------------------------------------------------------------------------
// k3: consolidated = w @ x  — 256 row-chunk partials (deterministic)
// ---------------------------------------------------------------------------
__global__ void __launch_bounds__(256)
k3_consolidate(const float* __restrict__ x, const float* __restrict__ logits,
               const int* __restrict__ evflags, const float* __restrict__ stats,
               float* __restrict__ partial) {
  __shared__ float wbuf[64];
  int tid = threadIdx.x;
  int rbase = blockIdx.x * 64;
  if (tid < 64) {
    int row = rbase + tid;
    wbuf[tid] = evflags[row] ? __expf(logits[row] - stats[0]) * stats[1] : 0.0f;
  }
  __syncthreads();
  float acc[8];
#pragma unroll
  for (int j = 0; j < 8; ++j) acc[j] = 0.0f;
  for (int r = 0; r < 64; ++r) {
    float wv = wbuf[r];                       // block-uniform -> cheap skip
    if (wv != 0.0f) {
      const float* xr = x + (size_t)(rbase + r) * HSZ;
#pragma unroll
      for (int j = 0; j < 8; ++j) acc[j] += wv * xr[tid + j * 256];
    }
  }
  float* pr = partial + (size_t)blockIdx.x * HSZ;
#pragma unroll
  for (int j = 0; j < 8; ++j) pr[tid + j * 256] = acc[j];
}

__global__ void __launch_bounds__(256)
k3b_reduce(const float* __restrict__ partial, float* __restrict__ consolidated) {
  int h = blockIdx.x * 256 + threadIdx.x;     // 2048
  float s = 0.0f;
  for (int b = 0; b < 256; ++b) s += partial[(size_t)b * HSZ + h];
  consolidated[h] = s;
}

// ---------------------------------------------------------------------------
// k4a/k4b: retrieval MLP on the single consolidated vector (coalesced)
// ---------------------------------------------------------------------------
__global__ void __launch_bounds__(256)
k4a_r1(const float* __restrict__ consolidated, const float* __restrict__ Wr1,
       const float* __restrict__ br1, float* __restrict__ hr) {
  int j = blockIdx.x * 256 + threadIdx.x;     // 1024
  float acc = br1[j];
  for (int k = 0; k < HSZ; ++k) acc += consolidated[k] * Wr1[(size_t)k * H2 + j];
  hr[j] = fmaxf(acc, 0.0f);
}

__global__ void __launch_bounds__(256)
k4b_r2(const float* __restrict__ hr, const float* __restrict__ Wr2,
       const float* __restrict__ br2, const int* __restrict__ hasEvents,
       float* __restrict__ addv) {
  int h = blockIdx.x * 256 + threadIdx.x;     // 2048
  float acc = br2[h];
  for (int j = 0; j < H2; ++j) acc += hr[j] * Wr2[(size_t)j * HSZ + h];
  float s = 1.0f / (1.0f + __expf(-acc));
  addv[h] = (*hasEvents) ? s : 0.0f;
}

// ---------------------------------------------------------------------------
// k5: out = x + add[None, :]   (float4 streaming, HBM-bound)
// ---------------------------------------------------------------------------
__global__ void __launch_bounds__(256)
k5_add(const float* __restrict__ x, const float* __restrict__ addv,
       float* __restrict__ out) {
  size_t i4 = ((size_t)blockIdx.x * 256 + threadIdx.x) * 4;
  int h = (int)(i4 & (HSZ - 1));
  float4 xv = *(const float4*)(x + i4);
  float4 av = *(const float4*)(addv + h);
  float4 o = make_float4(xv.x + av.x, xv.y + av.y, xv.z + av.z, xv.w + av.w);
  *(float4*)(out + i4) = o;
}

// ---------------------------------------------------------------------------
extern "C" void kernel_launch(void* const* d_in, const int* in_sizes, int n_in,
                              void* d_out, int out_size, void* d_ws, size_t ws_size,
                              hipStream_t stream) {
  const float* x    = (const float*)d_in[0];
  const float* imp  = (const float*)d_in[1];
  const float* We1  = (const float*)d_in[2];
  const float* be1  = (const float*)d_in[3];
  const float* We2  = (const float*)d_in[4];
  const float* be2  = (const float*)d_in[5];
  const float* Wr1  = (const float*)d_in[6];
  const float* br1  = (const float*)d_in[7];
  const float* Wr2  = (const float*)d_in[8];
  const float* br2  = (const float*)d_in[9];
  float* out = (float*)d_out;

  char* ws = (char*)d_ws;
  size_t off = 0;
  auto alloc = [&](size_t bytes) { char* p = ws + off; off = (off + bytes + 255) & ~(size_t)255; return p; };

  __bf16* B1      = (__bf16*)alloc((size_t)H2 * HSZ * 2);   // 4 MB  bf16 W_e1^T
  float*  logits  = (float*) alloc((size_t)NB * 4);
  int*    evflags = (int*)   alloc((size_t)NB * 4);
  int*    hasEv   = (int*)   alloc(256);
  float*  stats   = (float*) alloc(256);
  float*  partial = (float*) alloc((size_t)256 * HSZ * 4);  // 2 MB
  float*  cons    = (float*) alloc((size_t)HSZ * 4);
  float*  hr      = (float*) alloc((size_t)H2 * 4);
  float*  addv    = (float*) alloc((size_t)HSZ * 4);

  k0_prep<<<(H2 * HSZ) / 256, 256, 0, stream>>>(We1, B1, hasEv);
  k1_detect<<<NB / 16, 256, 0, stream>>>(x, B1, be1, We2, be2, imp,
                                         logits, evflags, hasEv);
  k2_stats<<<1, 1024, 0, stream>>>(logits, stats);
  k3_consolidate<<<NB / 64, 256, 0, stream>>>(x, logits, evflags, stats, partial);
  k3b_reduce<<<HSZ / 256, 256, 0, stream>>>(partial, cons);
  k4a_r1<<<H2 / 256, 256, 0, stream>>>(cons, Wr1, br1, hr);
  k4b_r2<<<HSZ / 256, 256, 0, stream>>>(hr, Wr2, br2, hasEv, addv);
  k5_add<<<(size_t)NB * HSZ / 4 / 256, 256, 0, stream>>>(x, addv, out);
}
